// Model_18502719111625
// MI455X (gfx1250) — matrix operand order repaired and checked by reading, not compiled, not run on hardware
//
#include <hip/hip_runtime.h>
#include <hip/hip_bf16.h>

// ---------------------------------------------------------------------------
// BRITS forward for gfx1250 (MI455X): bf16 WMMA (16x16x32) recurrent scan
// with TDM (tensor_load_to_lds) double-buffered input prefetch.
// B=1024, T=100, D=64, H=128.
// ---------------------------------------------------------------------------
#define BB 1024
#define TT 100
#define DD 64
#define HH 128
#define IMPUTE_W 0.3f

typedef __attribute__((ext_vector_type(16))) __bf16   v16bf;
typedef __attribute__((ext_vector_type(8)))  float    v8f;
typedef __attribute__((ext_vector_type(4)))  unsigned u32x4;
typedef __attribute__((ext_vector_type(8)))  int      i32x8;
typedef __attribute__((ext_vector_type(4)))  int      i32x4;
typedef __attribute__((ext_vector_type(4)))  float    f32x4;

union BF16x16 { u32x4 q[2]; v16bf v; };

__device__ __forceinline__ unsigned short bfbits(float f) {
    return (unsigned short)(__builtin_bit_cast(unsigned int, f) >> 16);
}
__device__ __forceinline__ unsigned pk2(float lo, float hi) {
    return (__builtin_bit_cast(unsigned int, lo) >> 16) |
           (__builtin_bit_cast(unsigned int, hi) & 0xFFFF0000u);
}
__device__ __forceinline__ float sigf(float x) { return 1.0f / (1.0f + __expf(-x)); }
__device__ __forceinline__ float tanhfast(float x) {
    x = fminf(fmaxf(x, -15.0f), 15.0f);
    float e = __expf(2.0f * x);
    return (e - 1.0f) / (e + 1.0f);
}

__device__ __forceinline__ v8f wmma_bf16(v16bf a, v16bf b, v8f c) {
    return __builtin_amdgcn_wmma_f32_16x16x32_bf16(
        false, a, false, b, (short)0, c, false, false);
}

// A operand (16x32 bf16) from a row-major f32 LDS tile (row stride 64 floats).
// ISA A layout: lane holds row = lane&15; K runs {cb..cb+7, cb+16..cb+23}.
__device__ __forceinline__ v16bf ldsAf32(const float* st, int lane, int cb) {
    const float* p = st + (lane & 15) * DD + cb;
    f32x4 a0 = *(const f32x4*)(p);
    f32x4 a1 = *(const f32x4*)(p + 4);
    f32x4 b0 = *(const f32x4*)(p + 16);
    f32x4 b1 = *(const f32x4*)(p + 20);
    BF16x16 u;
    u.q[0] = (u32x4){pk2(a0.x, a0.y), pk2(a0.z, a0.w), pk2(a1.x, a1.y), pk2(a1.z, a1.w)};
    u.q[1] = (u32x4){pk2(b0.x, b0.y), pk2(b0.z, b0.w), pk2(b1.x, b1.y), pk2(b1.z, b1.w)};
    return u.v;
}

// A operand from a row-major bf16 LDS staging area (row stride W ushorts).
__device__ __forceinline__ v16bf ldsA(const unsigned short* st, int W, int lane, int cb) {
    const unsigned short* p = st + (lane & 15) * W + cb;
    BF16x16 u;
    u.q[0] = *(const u32x4*)p;
    u.q[1] = *(const u32x4*)(p + 16);
    return u.v;
}

// B operand from the pre-swizzled weight buffer: 32B contiguous per lane.
// (Loop-invariant -> compiler hoists all tiles into VGPRs for the scan.)
__device__ __forceinline__ v16bf loadB(const unsigned short* sw, int tile, int lane) {
    const unsigned short* p = sw + tile * 512 + lane * 16;
    BF16x16 u;
    u.q[0] = *(const u32x4*)p;
    u.q[1] = *(const u32x4*)(p + 8);
    return u.v;
}

// ---------------------------------------------------------------------------
// TDM: async-load one 2D tile [16 rows x 64 f32], row stride 6400 f32, into LDS.
// D# per CDNA5 ISA ch.8: group0 = {count|flags, lds_addr, global_addr, type=2},
// group1 = {data_size, tensor dims, tile dims, strides}. Tracked by TENSORcnt.
// clang-23 toolchain: 6-arg builtin (g0, g1, g2, g3, g4, cpol).
// ---------------------------------------------------------------------------
__device__ __forceinline__ void tdm_load_tile(const float* gptr, unsigned lds_byte) {
    unsigned long long ga = (unsigned long long)(size_t)gptr;
    u32x4 g0;
    g0.x = 1u;                                          // count=1, user desc
    g0.y = lds_byte;                                    // lds_addr
    g0.z = (unsigned)(ga & 0xFFFFFFFFu);                // global_addr[31:0]
    g0.w = (unsigned)((ga >> 32) & 0x1FFFFFFu)          // global_addr[56:32]
           | 0x80000000u;                               // type=2 ("image")
    i32x8 g1;
    g1[0] = (int)(2u << 16);        // wg_mask=0, data_size=2 (4B), no flags
    g1[1] = (int)(64u << 16);       // atomic_barrier_addr=0 | tensor_dim0.lo16=64
    g1[2] = (int)(16u << 16);       // tensor_dim0.hi16=0   | tensor_dim1.lo16=16
    g1[3] = (int)(64u << 16);       // tensor_dim1.hi16=0   | tile_dim0=64
    g1[4] = 16;                     // tile_dim1=16         | tile_dim2=0
    g1[5] = (int)(TT * DD);         // tensor_dim0_stride[31:0] = 6400
    g1[6] = 0;                      // stride0.hi16=0 | tensor_dim1_stride.lo16=0
    g1[7] = 0;
    i32x4 z4 = {0, 0, 0, 0};
    i32x8 z8 = {0, 0, 0, 0, 0, 0, 0, 0};
    __builtin_amdgcn_tensor_load_to_lds(g0, g1, z4, z4, z8, 0);
}

__device__ __forceinline__ unsigned lds_off(const void* p) {
    return (unsigned)(size_t)p;     // LDS aperture: addr[31:0] = LDS offset
}

// ---------------------------------------------------------------------------
// Prep: f32 -> bf16 weight conversion + swizzle into WMMA B-operand layout.
// B[k][n] = W[n][k]. Tile = 32K x 16N; lane l -> n = l&15, k = (l>>4)*8 + (e&7) + (e>>3)*16.
// ---------------------------------------------------------------------------
__device__ __forceinline__ void swz(const float* W, int K, unsigned short* dst, int idx) {
    int tile = idx >> 9;
    int rem  = idx & 511;
    int l = rem >> 4, e = rem & 15;
    int KC = K >> 5;
    int nt = tile / KC, kc = tile - nt * KC;
    int n = nt * 16 + (l & 15);
    int k = kc * 32 + (l >> 4) * 8 + (e & 7) + (e >> 3) * 16;
    dst[idx] = bfbits(W[n * K + k]);
}

__global__ void prep_weights(const float* Wdec, const float* Wreg,
                             const float* Wih, const float* Whh,
                             const float* b_ih, const float* b_hh,
                             unsigned short* swWd, unsigned short* swWr,
                             unsigned short* swWih, unsigned short* swWhh,
                             float* bias_g) {
    int id = blockIdx.x * blockDim.x + threadIdx.x;
    if (id < 8192)  { swz(Wdec,  64, swWd, id); return; }
    id -= 8192;
    if (id < 8192)  { swz(Wreg, 128, swWr, id); return; }
    id -= 8192;
    if (id < 65536) { swz(Wih, 128, swWih, id); return; }
    id -= 65536;
    if (id < 65536) { swz(Whh, 128, swWhh, id); return; }
    id -= 65536;
    if (id < 4 * HH) bias_g[id] = b_ih[id] + b_hh[id];
}

// Per-timestep loss denominator: sum(eval_masks[:,t,:]) + 1e-5. Also zeroes acc.
__global__ void prep_denom(const float* eval_masks, float* denom, float* loss_acc) {
    __shared__ float red[256];
    int t = blockIdx.x;
    float s = 0.0f;
    for (int i = threadIdx.x; i < BB * DD; i += 256) {
        int b = i >> 6, d = i & 63;
        s += eval_masks[(size_t)b * TT * DD + (size_t)t * DD + d];
    }
    red[threadIdx.x] = s;
    __syncthreads();
    for (int w = 128; w > 0; w >>= 1) {
        if (threadIdx.x < w) red[threadIdx.x] += red[threadIdx.x + w];
        __syncthreads();
    }
    if (threadIdx.x == 0) {
        denom[t] = red[0] + 1e-5f;
        if (t == 0) *loss_acc = 0.0f;
    }
}

// ---------------------------------------------------------------------------
// Main scan. 64 blocks x 128 threads (4 waves). Each block owns 16 batch rows;
// waves split the N dimension of every GEMM. h/c in f32 WMMA C layout.
// TDM prefetches step t+1's five input tiles while step t computes.
// ---------------------------------------------------------------------------
__global__ __launch_bounds__(128) void brits_main(
    const float* __restrict__ values, const float* __restrict__ masks,
    const float* __restrict__ deltas, const float* __restrict__ evals,
    const float* __restrict__ eval_masks,
    const float* __restrict__ b_decay, const float* __restrict__ b_reg,
    const unsigned short* __restrict__ swWd, const unsigned short* __restrict__ swWr,
    const unsigned short* __restrict__ swWih, const unsigned short* __restrict__ swWhh,
    const float* __restrict__ bias_g, const float* __restrict__ denom,
    float* loss_acc, float* d_out) {
    __shared__ __align__(16) unsigned short h_stage[16 * HH];     // h, bf16 row-major
    __shared__ __align__(16) unsigned short lstm_stage[16 * HH];  // [x_c | m], bf16
    __shared__ __align__(16) float in_stage[2][5][16 * DD];       // dbl-buf input tiles

    const int lane = threadIdx.x & 31;
    const int wave = threadIdx.x >> 5;      // 0..3
    const int col  = lane & 15;
    const int hi   = lane >> 4;             // C-layout half select
    const int batch0 = blockIdx.x * 16;
    const size_t TD = (size_t)TT * DD;

    v8f h[2], c[2];
    #pragma unroll
    for (int q = 0; q < 2; ++q)
        #pragma unroll
        for (int r = 0; r < 8; ++r) { h[q][r] = 0.0f; c[q][r] = 0.0f; }

    float lsum = 0.0f;
    float* imps = d_out + 1;

    // Prologue: TDM prefetch of step 0's tiles into buffer 0 (wave 0 issues).
    if (wave == 0) {
        size_t off = (size_t)batch0 * TD;
        tdm_load_tile(values     + off, lds_off(&in_stage[0][0][0]));
        tdm_load_tile(masks      + off, lds_off(&in_stage[0][1][0]));
        tdm_load_tile(deltas     + off, lds_off(&in_stage[0][2][0]));
        tdm_load_tile(evals      + off, lds_off(&in_stage[0][3][0]));
        tdm_load_tile(eval_masks + off, lds_off(&in_stage[0][4][0]));
    }

    for (int t = 0; t < TT; ++t) {
        const int cur = t & 1;
        // Wait for this step's TDM tiles (wave 0 tracks TENSORcnt), publish to
        // all waves; barrier also retires last step's LDS readers.
        if (wave == 0) __builtin_amdgcn_s_wait_tensorcnt(0);
        __syncthreads();

        // Kick off next step's TDM prefetch into the other buffer.
        if (wave == 0 && t + 1 < TT) {
            int nb = 1 - cur;
            size_t off = (size_t)batch0 * TD + (size_t)(t + 1) * DD;
            tdm_load_tile(values     + off, lds_off(&in_stage[nb][0][0]));
            tdm_load_tile(masks      + off, lds_off(&in_stage[nb][1][0]));
            tdm_load_tile(deltas     + off, lds_off(&in_stage[nb][2][0]));
            tdm_load_tile(evals      + off, lds_off(&in_stage[nb][3][0]));
            tdm_load_tile(eval_masks + off, lds_off(&in_stage[nb][4][0]));
        }

        const float* sV  = in_stage[cur][0];
        const float* sM  = in_stage[cur][1];
        const float* sD  = in_stage[cur][2];
        const float* sE  = in_stage[cur][3];
        const float* sEM = in_stage[cur][4];

        // ---- gamma = exp(-relu(d @ Wd.T + bd)); h *= gamma (owned n-tiles) ----
        v16bf Ad0 = ldsAf32(sD, lane, hi * 8);
        v16bf Ad1 = ldsAf32(sD, lane, 32 + hi * 8);
        #pragma unroll
        for (int q = 0; q < 2; ++q) {
            int n = wave * 2 + q;
            v8f acc = {};
            acc = wmma_bf16(Ad0, loadB(swWd, n * 2 + 0, lane), acc);
            acc = wmma_bf16(Ad1, loadB(swWd, n * 2 + 1, lane), acc);
            float bd = b_decay[n * 16 + col];
            #pragma unroll
            for (int r = 0; r < 8; ++r) {
                float g = fmaxf(acc[r] + bd, 0.0f);
                h[q][r] *= __expf(-g);
            }
        }
        #pragma unroll
        for (int q = 0; q < 2; ++q) {
            int n = wave * 2 + q;
            #pragma unroll
            for (int r = 0; r < 8; ++r)
                h_stage[(r + 8 * hi) * HH + n * 16 + col] = bfbits(h[q][r]);
        }
        __syncthreads();

        // ---- x_h = h @ Wr.T + br (wave owns one of 4 D-tiles); loss + x_c ----
        v8f xh = {};
        #pragma unroll
        for (int kc = 0; kc < 4; ++kc) {
            v16bf a = ldsA(h_stage, HH, lane, kc * 32 + hi * 8);
            xh = wmma_bf16(a, loadB(swWr, wave * 4 + kc, lane), xh);
        }
        {
            float br   = b_reg[wave * 16 + col];
            int   colg = wave * 16 + col;
            float part = 0.0f;
            size_t idx0 = (size_t)(batch0 + 8 * hi) * TD + (size_t)t * DD + colg;
            #pragma unroll
            for (int r = 0; r < 8; ++r) {
                int li = (r + 8 * hi) * DD + colg;
                float v  = xh[r] + br;
                float ev = sE[li], em = sEM[li];
                part += fabsf(v - ev) * em;
                float xv = sV[li], mv = sM[li];
                float xc = mv * xv + (1.0f - mv) * v;
                imps[idx0 + (size_t)r * TD] = v;
                lstm_stage[(r + 8 * hi) * HH + colg]      = bfbits(xc);
                lstm_stage[(r + 8 * hi) * HH + DD + colg] = bfbits(mv);
            }
            #pragma unroll
            for (int off = 16; off > 0; off >>= 1)
                part += __shfl_xor(part, off, 32);
            lsum += part / denom[t];
        }
        __syncthreads();

        // ---- gates = [x_c,m] @ Wih.T + h @ Whh.T + (b_ih+b_hh); LSTM update ----
        #pragma unroll
        for (int q = 0; q < 2; ++q) {
            int j = wave * 2 + q;   // H column block; matches owned h/c tiles
            v8f gi, gf, gg, go;
            float bi = bias_g[          j * 16 + col];
            float bf = bias_g[    HH +  j * 16 + col];
            float bg = bias_g[2 * HH +  j * 16 + col];
            float bo = bias_g[3 * HH +  j * 16 + col];
            #pragma unroll
            for (int r = 0; r < 8; ++r) { gi[r] = bi; gf[r] = bf; gg[r] = bg; go[r] = bo; }
            #pragma unroll
            for (int kc = 0; kc < 4; ++kc) {
                v16bf a = ldsA(lstm_stage, HH, lane, kc * 32 + hi * 8);
                gi = wmma_bf16(a, loadB(swWih, (j)      * 4 + kc, lane), gi);
                gf = wmma_bf16(a, loadB(swWih, (8  + j) * 4 + kc, lane), gf);
                gg = wmma_bf16(a, loadB(swWih, (16 + j) * 4 + kc, lane), gg);
                go = wmma_bf16(a, loadB(swWih, (24 + j) * 4 + kc, lane), go);
            }
            #pragma unroll
            for (int kc = 0; kc < 4; ++kc) {
                v16bf a = ldsA(h_stage, HH, lane, kc * 32 + hi * 8);
                gi = wmma_bf16(a, loadB(swWhh, (j)      * 4 + kc, lane), gi);
                gf = wmma_bf16(a, loadB(swWhh, (8  + j) * 4 + kc, lane), gf);
                gg = wmma_bf16(a, loadB(swWhh, (16 + j) * 4 + kc, lane), gg);
                go = wmma_bf16(a, loadB(swWhh, (24 + j) * 4 + kc, lane), go);
            }
            #pragma unroll
            for (int r = 0; r < 8; ++r) {
                float iv = sigf(gi[r]);
                float fv = sigf(gf[r]);
                float gv = tanhfast(gg[r]);
                float ov = sigf(go[r]);
                float cv = fv * c[q][r] + iv * gv;
                c[q][r] = cv;
                h[q][r] = ov * tanhfast(cv);
            }
        }
    }
    if (lane == 0) atomicAdd(loss_acc, lsum);
}

// Tuple outputs 3 & 4 are pass-through copies of evals / eval_masks.
__global__ void copy_tail(const float* __restrict__ evals,
                          const float* __restrict__ eval_masks,
                          float* __restrict__ out) {
    const size_t N = (size_t)BB * TT * DD;
    for (size_t i = (size_t)blockIdx.x * blockDim.x + threadIdx.x; i < 2 * N;
         i += (size_t)gridDim.x * blockDim.x)
        out[1 + N + i] = (i < N) ? evals[i] : eval_masks[i - N];
}

__global__ void finalize_loss(const float* loss_acc, float* out) {
    out[0] = IMPUTE_W * loss_acc[0];
}

extern "C" void kernel_launch(void* const* d_in, const int* in_sizes, int n_in,
                              void* d_out, int out_size, void* d_ws, size_t ws_size,
                              hipStream_t stream) {
    const float* values     = (const float*)d_in[0];
    const float* masks      = (const float*)d_in[1];
    const float* deltas     = (const float*)d_in[2];
    const float* evals      = (const float*)d_in[3];
    const float* eval_masks = (const float*)d_in[4];
    const float* W_decay    = (const float*)d_in[5];
    const float* b_decay    = (const float*)d_in[6];
    const float* W_reg      = (const float*)d_in[7];
    const float* b_reg      = (const float*)d_in[8];
    const float* W_ih       = (const float*)d_in[9];
    const float* W_hh       = (const float*)d_in[10];
    const float* b_ih       = (const float*)d_in[11];
    const float* b_hh       = (const float*)d_in[12];

    char* ws = (char*)d_ws;
    float*          loss_acc = (float*)(ws + 0);
    float*          denom    = (float*)(ws + 64);        // 100 floats
    float*          bias_g   = (float*)(ws + 512);       // 512 floats
    unsigned short* swWd     = (unsigned short*)(ws + 4096);                     // 16 KB
    unsigned short* swWr     = (unsigned short*)(ws + 4096 + 16384);             // 16 KB
    unsigned short* swWih    = (unsigned short*)(ws + 4096 + 32768);             // 128 KB
    unsigned short* swWhh    = (unsigned short*)(ws + 4096 + 32768 + 131072);    // 128 KB

    prep_denom<<<TT, 256, 0, stream>>>(eval_masks, denom, loss_acc);
    prep_weights<<<(8192 + 8192 + 65536 + 65536 + 4 * HH + 255) / 256, 256, 0, stream>>>(
        W_decay, W_reg, W_ih, W_hh, b_ih, b_hh, swWd, swWr, swWih, swWhh, bias_g);
    brits_main<<<BB / 16, 128, 0, stream>>>(
        values, masks, deltas, evals, eval_masks, b_decay, b_reg,
        swWd, swWr, swWih, swWhh, bias_g, denom, loss_acc, (float*)d_out);
    copy_tail<<<4096, 256, 0, stream>>>(evals, eval_masks, (float*)d_out);
    finalize_loss<<<1, 1, 0, stream>>>(loss_acc, (float*)d_out);
}
